// MemoryBank_56032143343998
// MI455X (gfx1250) — matrix operand order repaired and checked by reading, not compiled, not run on hardware
//
#include <hip/hip_runtime.h>
#include <hip/hip_bf16.h>
#include <math.h>

typedef __attribute__((ext_vector_type(16))) _Float16 v16h;
typedef __attribute__((ext_vector_type(8)))  float    v8f;

#define TOPK 20
#define NSPLIT 8

__device__ __forceinline__ float gelu_exact(float x) {
    return 0.5f * x * (1.0f + erff(x * 0.70710678118654752440f));
}

// Insert (vvl, n) into ascending sorted list of TOPK; Lv[0] is the minimum.
// Kept un-unrolled: this is the rare path, keep static code small.
__device__ __forceinline__ void topk_insert(float* Lv, int* Li, float vvl, int n) {
    int p = 0;
    #pragma clang loop unroll(disable)
    while (p < TOPK - 1 && Lv[p + 1] < vvl) { Lv[p] = Lv[p + 1]; Li[p] = Li[p + 1]; p++; }
    Lv[p] = vvl; Li[p] = n;
}

// ---------------------------------------------------------------------------
// k_prep: anorm[b][n] = sum_k km[b][k][n]^2 (zero-padded to nch*16),
//         denom[b][n] = HW, S zeroed.
// ---------------------------------------------------------------------------
__global__ void __launch_bounds__(256)
k_prep(const float* __restrict__ km, float* __restrict__ anorm,
       float* __restrict__ denom, float* __restrict__ S,
       int B, int NE, int nch, int HW) {
    int idx = blockIdx.x * blockDim.x + threadIdx.x;
    int NP = nch * 16;
    if (idx < B * NP) {
        int b = idx / NP, n = idx % NP;
        float s = 0.f;
        if (n < NE) {
            const float* p = km + (size_t)b * 64 * NE + n;
            #pragma unroll
            for (int k = 0; k < 64; k++) {
                float x = p[(size_t)k * NE];
                s = fmaf(x, x, s);
            }
            denom[(size_t)b * NE + n] = (float)HW;
        }
        anorm[idx] = s;
    }
    if (idx < B * 16) S[idx] = 0.f;
}

// ---------------------------------------------------------------------------
// k_swz: pre-swizzle km (fp32, [B,64,NE]) into f16 WMMA A-operand layout:
// kmsw[b][chunk][q][lane][j]  (chunk = 16 rows of n, q = K-half of 64)
// 16-bit A 16x32 layout: lane<16: j<8 -> K=q*32+j, j>=8 -> K=q*32+j+8
//                        lane>=16: +8 on K offset.  M (row) = lane&15.
// ---------------------------------------------------------------------------
__global__ void __launch_bounds__(256)
k_swz(const float* __restrict__ km, _Float16* __restrict__ kmsw,
      int B, int NE, int nch) {
    int idx = blockIdx.x * blockDim.x + threadIdx.x;
    int tot = B * nch * 1024;
    if (idx >= tot) return;
    int j  = idx & 15;
    int l  = (idx >> 4) & 31;
    int q  = (idx >> 9) & 1;
    int ch = (idx >> 10) % nch;
    int b  = idx / (nch * 1024);
    int n  = ch * 16 + (l & 15);
    int k  = q * 32 + ((l & 16) ? 8 : 0) + ((j < 8) ? j : (j + 8));
    float v = (n < NE) ? km[(size_t)b * 64 * NE + (size_t)k * NE + n] : 0.f;
    kmsw[idx] = (_Float16)v;
}

// ---------------------------------------------------------------------------
// k_cwscore: channel-weight MLP + layernorm, per-pixel score MLP,
//            softmax over HW -> posw[b][h].  One block per batch.
// ---------------------------------------------------------------------------
__global__ void __launch_bounds__(256)
k_cwscore(const float* __restrict__ vq,
          const float* __restrict__ mw1, const float* __restrict__ mb1,
          const float* __restrict__ mw2, const float* __restrict__ mb2,
          const float* __restrict__ lg,  const float* __restrict__ lb,
          const float* __restrict__ pw1, const float* __restrict__ pb1,
          const float* __restrict__ pw2, const float* __restrict__ pb2,
          float* __restrict__ posw, int HW, int sc) {
    __shared__ float red[256];
    __shared__ float pooled[16];
    __shared__ float cw[16];
    __shared__ float bc[2];
    const int tid = threadIdx.x;
    const int b = blockIdx.x;
    const float* v = vq + (size_t)b * 16 * HW;

    // pooled spatial mean per channel
    {
        int c = tid >> 4, t = tid & 15;
        float s = 0.f;
        for (int h = t; h < HW; h += 16) s += v[(size_t)c * HW + h];
        red[tid] = s;
        __syncthreads();
        if (t == 0) {
            float a = 0.f;
            #pragma unroll
            for (int i = 0; i < 16; i++) a += red[c * 16 + i];
            pooled[c] = a / (float)HW;
        }
        __syncthreads();
    }
    // 16->1->16 MLP + layernorm (tiny; serial on thread 0)
    if (tid == 0) {
        float hsum = mb1[sc];
        #pragma unroll
        for (int i = 0; i < 16; i++) hsum = fmaf(pooled[i], mw1[sc * 16 + i], hsum);
        float hid = gelu_exact(hsum);
        float x[16]; float m = 0.f;
        #pragma unroll
        for (int i = 0; i < 16; i++) { x[i] = fmaf(hid, mw2[sc * 16 + i], mb2[sc * 16 + i]); m += x[i]; }
        m *= (1.f / 16.f);
        float var = 0.f;
        #pragma unroll
        for (int i = 0; i < 16; i++) { float d = x[i] - m; var = fmaf(d, d, var); }
        var *= (1.f / 16.f);
        float r = rsqrtf(var + 1e-5f);
        #pragma unroll
        for (int i = 0; i < 16; i++) cw[i] = (x[i] - m) * r * lg[sc * 16 + i] + lb[sc * 16 + i];
    }
    __syncthreads();
    // per-pixel score: 16->8->1 MLP; track max for stable softmax
    float lmax = -INFINITY;
    for (int h = tid; h < HW; h += 256) {
        float f[16];
        #pragma unroll
        for (int c = 0; c < 16; c++) f[c] = v[(size_t)c * HW + h] * cw[c];
        float s = pb2[sc];
        #pragma unroll
        for (int j = 0; j < 8; j++) {
            float a = pb1[sc * 8 + j];
            #pragma unroll
            for (int c = 0; c < 16; c++) a = fmaf(f[c], pw1[(sc * 16 + c) * 8 + j], a);
            s = fmaf(gelu_exact(a), pw2[sc * 8 + j], s);
        }
        s *= 10.f;  // /TEMP, TEMP=0.1
        posw[(size_t)b * HW + h] = s;
        lmax = fmaxf(lmax, s);
    }
    red[tid] = lmax; __syncthreads();
    for (int st = 128; st > 0; st >>= 1) { if (tid < st) red[tid] = fmaxf(red[tid], red[tid + st]); __syncthreads(); }
    if (tid == 0) bc[0] = red[0];
    __syncthreads();
    float mx = bc[0];
    float lsum = 0.f;
    for (int h = tid; h < HW; h += 256) lsum += expf(posw[(size_t)b * HW + h] - mx);
    red[tid] = lsum; __syncthreads();
    for (int st = 128; st > 0; st >>= 1) { if (tid < st) red[tid] += red[tid + st]; __syncthreads(); }
    if (tid == 0) bc[1] = red[0];
    __syncthreads();
    float inv = 1.f / bc[1];
    for (int h = tid; h < HW; h += 256) {
        size_t p = (size_t)b * HW + h;
        posw[p] = expf(posw[p] - mx) * inv;
    }
}

// ---------------------------------------------------------------------------
// k_afftopk: fused affinity GEMM (f16 WMMA) + per-column top-20 + softmax.
// Block = NSPLIT waves x 32 lanes. Block owns 32 columns (h); each wave takes
// an interleaved subset of NE chunks; lane L owns column h0+L.
// aff[n][h] = dot(km[:,n], qk[:,h])/4 - anorm[n]/8; -anorm/2 folded into C
// init so candidate value = acc*0.25.  Fast path: 15 v_max + 1 cmp per chunk.
// ---------------------------------------------------------------------------
__global__ void __launch_bounds__(32 * NSPLIT)
k_afftopk(const _Float16* __restrict__ kmsw, const float* __restrict__ anorm,
          const float* __restrict__ kq,
          int* __restrict__ tki, float* __restrict__ tkw,
          int NE, int nch, int HW) {
    const int b = blockIdx.y;
    const int h0 = blockIdx.x * 32;
    const int wv = threadIdx.x >> 5;
    const int lane = threadIdx.x & 31;

    __shared__ float tv[NSPLIT][32][TOPK];
    __shared__ int   ti[NSPLIT][32][TOPK];
    #pragma unroll
    for (int j = 0; j < TOPK; j++) { tv[wv][lane][j] = -INFINITY; ti[wv][lane][j] = 0; }

    // Load B operands (qk columns) once per wave. 16-bit B 32x16 layout:
    // lanes 0-15: N=lane, K=q*32+(j<8?j:j+8) ; lanes 16-31: N=lane-16, +8 on K.
    const float* qk = kq + (size_t)b * 64 * HW;
    v16h bv[2][2];
    const int colb = h0 + (lane & 15);
    const int koff = (lane & 16);
    #pragma unroll
    for (int t2 = 0; t2 < 2; t2++) {
        int h = colb + t2 * 16; if (h >= HW) h = HW - 1;
        #pragma unroll
        for (int q = 0; q < 2; q++) {
            #pragma unroll
            for (int j = 0; j < 16; j++) {
                int k = q * 32 + (koff >> 1) + ((j < 8) ? j : (j + 8));
                bv[t2][q][j] = (_Float16)qk[(size_t)k * HW + h];
            }
        }
    }

    float minv4 = -INFINITY;  // 4 * (current 20th-best aff value)
    const _Float16* kma = kmsw + (size_t)b * nch * 1024;
    const float* anb = anorm + (size_t)b * nch * 16;
    const bool lo = (lane < 16);

    for (int ch = wv; ch < nch; ch += NSPLIT) {
        if (ch + NSPLIT < nch)
            __builtin_prefetch(kma + (size_t)(ch + NSPLIT) * 1024, 0, 1);
        const v16h* ap = (const v16h*)(kma + (size_t)ch * 1024);
        v16h a0 = ap[lane];
        v16h a1 = ap[32 + lane];
        // C init = -anorm/2 for this lane's rows (lane<16: m=r; lane>=16: m=8+r)
        const float4* an4 = (const float4*)(anb + (size_t)ch * 16 + ((lane & 16) ? 8 : 0));
        float4 A0 = an4[0], A1 = an4[1];
        v8f cc;
        cc[0] = -0.5f * A0.x; cc[1] = -0.5f * A0.y; cc[2] = -0.5f * A0.z; cc[3] = -0.5f * A0.w;
        cc[4] = -0.5f * A1.x; cc[5] = -0.5f * A1.y; cc[6] = -0.5f * A1.z; cc[7] = -0.5f * A1.w;
        v8f c0 = cc, c1 = cc;
        c0 = __builtin_amdgcn_wmma_f32_16x16x32_f16(false, a0, false, bv[0][0], (short)0, c0, false, false);
        c0 = __builtin_amdgcn_wmma_f32_16x16x32_f16(false, a1, false, bv[0][1], (short)0, c0, false, false);
        c1 = __builtin_amdgcn_wmma_f32_16x16x32_f16(false, a0, false, bv[1][0], (short)0, c1, false, false);
        c1 = __builtin_amdgcn_wmma_f32_16x16x32_f16(false, a1, false, bv[1][1], (short)0, c1, false, false);

        // Give each lane its full 16-row column.  Each lane sends the value
        // its xor-16 partner needs (lo sends c1, hi sends c0): 8 bpermutes.
        float cand[16];
        #pragma unroll
        for (int r = 0; r < 8; r++) {
            float send = lo ? c1[r] : c0[r];
            float recv = __shfl_xor(send, 16, 32);
            cand[r]     = lo ? c0[r] : recv;   // rows m=0..7
            cand[8 + r] = lo ? recv  : c1[r];  // rows m=8..15
        }
        // Fast filter: does any candidate beat the current 20th-best?
        float cmax = cand[0];
        #pragma unroll
        for (int m = 1; m < 16; m++) cmax = fmaxf(cmax, cand[m]);
        if (cmax > minv4) {
            const int n0 = ch * 16;
            #pragma unroll
            for (int m = 0; m < 16; m++) {
                int n = n0 + m;
                if (cand[m] > minv4 && n < NE) {
                    topk_insert(tv[wv][lane], ti[wv][lane], cand[m] * 0.25f, n);
                    minv4 = tv[wv][lane][0] * 4.f;
                }
            }
        }
    }
    __syncthreads();

    // Wave 0 merges the NSPLIT per-wave lists and applies softmax over top-20
    if (threadIdx.x < 32) {
        const int L = threadIdx.x;
        const int h = h0 + L;
        if (h < HW) {
            float* Mv = tv[0][L]; int* Mi = ti[0][L];
            float mv0 = Mv[0];
            #pragma clang loop unroll(disable)
            for (int w = 1; w < NSPLIT; w++) {
                #pragma clang loop unroll(disable)
                for (int j = 0; j < TOPK; j++) {
                    float vvl = tv[w][L][j]; int n = ti[w][L][j];
                    if (vvl > mv0) {
                        topk_insert(Mv, Mi, vvl, n);
                        mv0 = Mv[0];
                    }
                }
            }
            float mx = Mv[TOPK - 1];
            float e[TOPK]; float sum = 0.f;
            #pragma unroll
            for (int j = 0; j < TOPK; j++) { e[j] = expf(Mv[j] - mx); sum += e[j]; }
            float inv = 1.f / sum;
            size_t base = ((size_t)b * HW + h) * TOPK;
            #pragma unroll
            for (int j = 0; j < TOPK; j++) { tkw[base + j] = e[j] * inv; tki[base + j] = Mi[j]; }
        }
    }
}

// ---------------------------------------------------------------------------
// k_scatter: ew = expm1(w * posw); denom[n] += ew (sparse scatter).
// ---------------------------------------------------------------------------
__global__ void __launch_bounds__(256)
k_scatter(const int* __restrict__ tki, const float* __restrict__ tkw,
          const float* __restrict__ posw, float* __restrict__ denom,
          float* __restrict__ ew, int B, int HW, int NE) {
    int idx = blockIdx.x * blockDim.x + threadIdx.x;
    if (idx >= B * HW * TOPK) return;
    int h = (idx / TOPK) % HW;
    int b = idx / (TOPK * HW);
    float e = expm1f(tkw[idx] * posw[(size_t)b * HW + h]);
    ew[idx] = e;
    atomicAdd(&denom[(size_t)b * NE + tki[idx]], e);
}

// ---------------------------------------------------------------------------
// k_vmd: vmd[b][n][c] = vm[b][c][n]/denom[b][n] (transpose);  S[b][c] = sum_n.
// ---------------------------------------------------------------------------
__global__ void __launch_bounds__(256)
k_vmd(const float* __restrict__ vm, const float* __restrict__ denom,
      float* __restrict__ vmd, float* __restrict__ S, int NE) {
    __shared__ float sacc[16];
    const int b = blockIdx.y;
    const int tid = threadIdx.x;
    if (tid < 16) sacc[tid] = 0.f;
    __syncthreads();
    int n = blockIdx.x * blockDim.x + tid;
    if (n < NE) {
        float inv = 1.f / denom[(size_t)b * NE + n];
        const float* vp = vm + (size_t)b * 16 * NE + n;
        float* op = vmd + ((size_t)b * NE + n) * 16;
        #pragma unroll
        for (int c = 0; c < 16; c++) {
            float x = vp[(size_t)c * NE] * inv;
            op[c] = x;
            atomicAdd(&sacc[c], x);
        }
    }
    __syncthreads();
    if (tid < 16) atomicAdd(&S[b * 16 + tid], sacc[tid]);
}

// ---------------------------------------------------------------------------
// k_final: out[b][c][h] = S[b][c] + sum_j ew[j]*vmd[b][idx_j][c]  (c<16)
//          out[b][16+c][h] = vq[b][c][h]
// ---------------------------------------------------------------------------
__global__ void __launch_bounds__(256)
k_final(const float* __restrict__ vq, const float* __restrict__ vmd,
        const float* __restrict__ S, const int* __restrict__ tki,
        const float* __restrict__ ew, float* __restrict__ out,
        int B, int HW, int NE) {
    int idx = blockIdx.x * blockDim.x + threadIdx.x;
    if (idx >= B * HW) return;
    int b = idx / HW, h = idx % HW;
    float acc[16];
    const float* sp = S + b * 16;
    #pragma unroll
    for (int c = 0; c < 16; c++) acc[c] = sp[c];
    size_t base = (size_t)idx * TOPK;
    #pragma unroll 4
    for (int j = 0; j < TOPK; j++) {
        int n = tki[base + j];
        float e = ew[base + j];
        const float* r = vmd + ((size_t)b * NE + n) * 16;
        #pragma unroll
        for (int c = 0; c < 16; c++) acc[c] = fmaf(e, r[c], acc[c]);
    }
    float* o = out + (size_t)b * 32 * HW + h;
    const float* v = vq + (size_t)b * 16 * HW + h;
    #pragma unroll
    for (int c = 0; c < 16; c++) o[(size_t)c * HW] = acc[c];
    #pragma unroll
    for (int c = 0; c < 16; c++) o[(size_t)(16 + c) * HW] = v[(size_t)c * HW];
}

// ---------------------------------------------------------------------------
extern "C" void kernel_launch(void* const* d_in, const int* in_sizes, int n_in,
                              void* d_out, int out_size, void* d_ws, size_t ws_size,
                              hipStream_t stream) {
    (void)in_sizes; (void)n_in; (void)out_size; (void)ws_size;
    const int B = 2;
    const float* mw1 = (const float*)d_in[12];
    const float* mb1 = (const float*)d_in[13];
    const float* mw2 = (const float*)d_in[14];
    const float* mb2 = (const float*)d_in[15];
    const float* lg  = (const float*)d_in[16];
    const float* lb  = (const float*)d_in[17];
    const float* pw1 = (const float*)d_in[18];
    const float* pb1 = (const float*)d_in[19];
    const float* pw2 = (const float*)d_in[20];
    const float* pb2 = (const float*)d_in[21];

    char* ws = (char*)d_ws;
    size_t off = 0;
    auto alloc = [&](size_t bytes) -> char* {
        off = (off + 255) & ~(size_t)255;
        char* p = ws + off;
        off += bytes;
        return p;
    };

    float* outp = (float*)d_out;
    size_t obase = 0;
    const int Hs[3] = {44, 22, 11};
    for (int i = 0; i < 3; ++i) {
        const int H = Hs[i], HW = H * H, NE = 8 * HW, nch = (NE + 15) / 16;
        const float* kq = (const float*)d_in[4 * i + 0];
        const float* vq = (const float*)d_in[4 * i + 1];
        const float* km = (const float*)d_in[4 * i + 2];
        const float* vm = (const float*)d_in[4 * i + 3];

        _Float16* kmsw = (_Float16*)alloc((size_t)B * nch * 1024 * 2);
        float* anorm = (float*)alloc((size_t)B * nch * 16 * 4);
        float* posw  = (float*)alloc((size_t)B * HW * 4);
        int*   tki   = (int*)  alloc((size_t)B * HW * TOPK * 4);
        float* tkw   = (float*)alloc((size_t)B * HW * TOPK * 4);
        float* ewb   = (float*)alloc((size_t)B * HW * TOPK * 4);
        float* denom = (float*)alloc((size_t)B * NE * 4);
        float* vmd   = (float*)alloc((size_t)B * NE * 16 * 4);
        float* S     = (float*)alloc((size_t)B * 16 * 4);

        int tp = B * nch * 16;
        k_prep<<<(tp + 255) / 256, 256, 0, stream>>>(km, anorm, denom, S, B, NE, nch, HW);
        int ts = B * nch * 1024;
        k_swz<<<(ts + 255) / 256, 256, 0, stream>>>(km, kmsw, B, NE, nch);
        k_cwscore<<<B, 256, 0, stream>>>(vq, mw1, mb1, mw2, mb2, lg, lb,
                                         pw1, pb1, pw2, pb2, posw, HW, i);
        dim3 ga((HW + 31) / 32, B);
        k_afftopk<<<ga, 32 * NSPLIT, 0, stream>>>(kmsw, anorm, kq, tki, tkw, NE, nch, HW);
        int tt = B * HW * TOPK;
        k_scatter<<<(tt + 255) / 256, 256, 0, stream>>>(tki, tkw, posw, denom, ewb, B, HW, NE);
        dim3 gv((NE + 255) / 256, B);
        k_vmd<<<gv, 256, 0, stream>>>(vm, denom, vmd, S, NE);
        int tf = B * HW;
        k_final<<<(tf + 255) / 256, 256, 0, stream>>>(vq, vmd, S, tki, ewb,
                                                      outp + obase, B, HW, NE);
        obase += (size_t)B * 32 * HW;
    }
}
